// SoftWindowBiMamba_50740743635361
// MI455X (gfx1250) — matrix operand order repaired and checked by reading, not compiled, not run on hardware
//
#include <hip/hip_runtime.h>
#include <hip/hip_bf16.h>

// ---------------------------------------------------------------------------
// Problem constants (from reference)
// ---------------------------------------------------------------------------
#define D_MODEL   1024
#define D_STATE   64
#define D_CONV    4
#define HEADDIM   64
#define D_INNER   2048                    // EXPAND * D_MODEL
#define NHEADS    32                      // D_INNER / HEADDIM
#define CONV_DIM  (D_INNER + 2 * D_STATE) // 2176
#define D_IN_PROJ (2 * D_INNER + 2 * D_STATE + NHEADS) // 4256
#define BATCH     4
#define SEQLEN    2048
#define NTOK      (BATCH * SEQLEN)        // 8192
#define EPS_RMS   1e-5f

typedef __attribute__((ext_vector_type(16))) __bf16 v16bf;
typedef __attribute__((ext_vector_type(8)))  float  v8f;

__device__ __forceinline__ float silu_f(float v) { return v / (1.f + __expf(-v)); }
__device__ __forceinline__ float softplus_f(float v) {
    return (v > 20.f) ? v : __logf(1.f + __expf(v));
}

// ---------------------------------------------------------------------------
// Async global->LDS staging (GLOBAL_LOAD_ASYNC_TO_LDS_B128, ASYNCcnt) when the
// toolchain exposes the gfx1250 builtins; otherwise fall back to b128 loads +
// ds stores. Builtin signature (from hipcc diagnostics): param0 is a pointer
// to GCC-style int4 in AS1, param1 int4 in AS3, then imm offset / imm cpol.
// LDS offset = low 32 bits of the flat address (ISA aperture rule).
// ---------------------------------------------------------------------------
#if defined(__has_builtin)
#if __has_builtin(__builtin_amdgcn_global_load_async_to_lds_b128) && \
    __has_builtin(__builtin_amdgcn_s_wait_asynccnt)
#define USE_ASYNC_LDS 1
#endif
#endif

#ifdef USE_ASYNC_LDS
typedef int vi4 __attribute__((vector_size(16)));
typedef __attribute__((address_space(1))) vi4 gvi4; // global int4
typedef __attribute__((address_space(3))) vi4 lvi4; // LDS int4
__device__ __forceinline__ void async_copy16(const void* g, void* l) {
    __builtin_amdgcn_global_load_async_to_lds_b128(
        (gvi4*)(unsigned long long)g,
        (lvi4*)(unsigned int)(unsigned long long)l, 0, 0);
}
#define ASYNC_WAIT() __builtin_amdgcn_s_wait_asynccnt(0)
#else
#define ASYNC_WAIT() ((void)0)
#endif

// ---------------------------------------------------------------------------
// f32 -> bf16 converts
// ---------------------------------------------------------------------------
__global__ void cvt_bf16_kernel(const float* __restrict__ in, __bf16* __restrict__ out, long n) {
    long i = (long)blockIdx.x * blockDim.x + threadIdx.x;
    if (i < n) out[i] = (__bf16)in[i];
}

// x (B,L,D) -> bf16 with L flipped per batch (for the backward direction)
__global__ void cvt_bf16_flipL_kernel(const float* __restrict__ in, __bf16* __restrict__ out) {
    long i = (long)blockIdx.x * blockDim.x + threadIdx.x;
    if (i >= (long)NTOK * D_MODEL) return;
    int  d   = (int)(i % D_MODEL);
    long row = i / D_MODEL;
    int  l   = (int)(row % SEQLEN);
    long b0  = row - l; // b * SEQLEN
    out[(b0 + (SEQLEN - 1 - l)) * (long)D_MODEL + d] = (__bf16)in[i];
}

// ---------------------------------------------------------------------------
// WMMA GEMM: C[M,N] = A[M,K] * B[N,K]^T   (bf16 in, f32 out)
// Block tile 128x128, 256 threads = 8 waves (2x4), wave tile 64x32,
// K-step 32, double-buffered LDS: one barrier per K-step, staging of step
// k+1 overlaps the 8 v_wmma_f32_16x16x32_bf16 of step k.
// Requires M%128==0, N%128==0, K%32==0 (GEMM1's N = 4256 is not a multiple of 128).
// ---------------------------------------------------------------------------
#define GEMM_BM 128
#define GEMM_BN 128
#define GEMM_BK 32
#define LDS_STR 40 // ushorts per LDS row (32 data + 8 pad): 16B-aligned rows,
                   // 20-dword row pitch -> conflict-free fragment gathers

__device__ __forceinline__ void stage_tile(const __bf16* __restrict__ A,
                                           const __bf16* __restrict__ Bw,
                                           unsigned short* sA, unsigned short* sB,
                                           int bm, int bn, int K, int k0, int tid) {
#pragma unroll
    for (int it = 0; it < 2; ++it) {
        int c    = tid + it * 256;   // chunk 0..511
        int row  = c >> 2;           // 0..127
        int col8 = (c & 3) * 8;      // 0,8,16,24
        const __bf16* ga = A + (long)(bm + row) * K + k0 + col8;
        const __bf16* gb = Bw + (long)(bn + row) * K + k0 + col8;
        unsigned short* la = sA + row * LDS_STR + col8;
        unsigned short* lb = sB + row * LDS_STR + col8;
#ifdef USE_ASYNC_LDS
        async_copy16(ga, la);
        async_copy16(gb, lb);
#else
        uint4 va = *reinterpret_cast<const uint4*>(ga);
        unsigned int* da = reinterpret_cast<unsigned int*>(la);
        da[0] = va.x; da[1] = va.y; da[2] = va.z; da[3] = va.w;
        uint4 vb = *reinterpret_cast<const uint4*>(gb);
        unsigned int* db = reinterpret_cast<unsigned int*>(lb);
        db[0] = vb.x; db[1] = vb.y; db[2] = vb.z; db[3] = vb.w;
#endif
    }
}

__global__ __launch_bounds__(256)
void gemm_bf16_wmma_kernel(const __bf16* __restrict__ A, const __bf16* __restrict__ Bw,
                           float* __restrict__ C, int M, int N, int K) {
    __shared__ unsigned short sA[2][GEMM_BM * LDS_STR];
    __shared__ unsigned short sB[2][GEMM_BN * LDS_STR];

    const int tid  = threadIdx.x;
    const int lane = tid & 31;
    const int wave = tid >> 5;
    const int wm   = wave >> 2;   // 0..1 : 64-row strip
    const int wn   = wave & 3;    // 0..3 : 32-col strip
    const int hl   = lane >> 4;   // lane half (ISA 16-bit WMMA layout)
    const int lcol = lane & 15;
    const int bm   = blockIdx.y * GEMM_BM;
    const int bn   = blockIdx.x * GEMM_BN;

    v8f acc[4][2];
#pragma unroll
    for (int mi = 0; mi < 4; ++mi)
#pragma unroll
        for (int ni = 0; ni < 2; ++ni)
#pragma unroll
            for (int e = 0; e < 8; ++e) acc[mi][ni][e] = 0.f;

    union Frag { unsigned int u[8]; v16bf v; };

    // Prologue: stage K-panel 0 into buffer 0.
    stage_tile(A, Bw, sA[0], sB[0], bm, bn, K, 0, tid);
    int buf = 0;

    for (int k0 = 0; k0 < K; k0 += GEMM_BK) {
        ASYNC_WAIT();          // my async writes into sX[buf] have landed
        __syncthreads();       // everyone's writes landed; prior reads of buf^1 done
        if (k0 + GEMM_BK < K) {
            stage_tile(A, Bw, sA[buf ^ 1], sB[buf ^ 1], bm, bn, K, k0 + GEMM_BK, tid);
#ifndef USE_ASYNC_LDS
            // keep a prefetch on the panel after next (global_prefetch_b8 path)
            if (k0 + 2 * GEMM_BK < K) {
                int row  = tid >> 2;
                int col8 = (tid & 3) * 8;
                __builtin_prefetch(A + (long)(bm + row) * K + k0 + 2 * GEMM_BK + col8, 0, 3);
                __builtin_prefetch(Bw + (long)(bn + row) * K + k0 + 2 * GEMM_BK + col8, 0, 3);
            }
#endif
        }

        const unsigned short* cA = sA[buf];
        const unsigned short* cB = sB[buf];

        // B fragments: 32x16 (KxN) bf16. lanes 0-15 hold K=0..15, lanes 16-31
        // hold K=16..31, VGPR v holds K=2v,2v+1; N = lane&15.
        Frag bfr[2];
#pragma unroll
        for (int ni = 0; ni < 2; ++ni) {
            int nrow = wn * 32 + ni * 16 + lcol;
#pragma unroll
            for (int v = 0; v < 8; ++v) {
                int k = hl * 16 + 2 * v;
                bfr[ni].u[v] = *reinterpret_cast<const unsigned int*>(&cB[nrow * LDS_STR + k]);
            }
        }
        // A fragments: 16x32 (MxK) bf16: VGPR v holds K=hl*16+2v,+1
        // (same k order as B); M = lane&15.
#pragma unroll
        for (int mi = 0; mi < 4; ++mi) {
            Frag af;
            int mrow = wm * 64 + mi * 16 + lcol;
#pragma unroll
            for (int v = 0; v < 8; ++v) {
                int k = hl * 16 + 2 * v;
                af.u[v] = *reinterpret_cast<const unsigned int*>(&cA[mrow * LDS_STR + k]);
            }
#pragma unroll
            for (int ni = 0; ni < 2; ++ni) {
                acc[mi][ni] = __builtin_amdgcn_wmma_f32_16x16x32_bf16(
                    false, af.v, false, bfr[ni].v, (short)0, acc[mi][ni], false, false);
            }
        }
        buf ^= 1;
    }

    // Store C: f32 16x16 layout: lane half hl, col = lane&15, VGPR r -> M = r + 8*hl.
#pragma unroll
    for (int mi = 0; mi < 4; ++mi)
#pragma unroll
        for (int ni = 0; ni < 2; ++ni) {
            int coln = bn + wn * 32 + ni * 16 + lcol;
#pragma unroll
            for (int r = 0; r < 8; ++r) {
                int rowm = bm + wm * 64 + mi * 16 + r + 8 * hl;
                C[(long)rowm * N + coln] = acc[mi][ni][r];
            }
        }
}

// ---------------------------------------------------------------------------
// Depthwise causal conv (width 4) + bias + SiLU over the xBC slice of zxbcdt.
// ---------------------------------------------------------------------------
__global__ void conv_silu_kernel(const float* __restrict__ zx, const float* __restrict__ cw,
                                 const float* __restrict__ cb, float* __restrict__ xconv) {
    long i = (long)blockIdx.x * blockDim.x + threadIdx.x;
    if (i >= (long)NTOK * CONV_DIM) return;
    int  c    = (int)(i % CONV_DIM);
    long row  = i / CONV_DIM;
    int  l    = (int)(row % SEQLEN);
    long brow = row - l; // b * SEQLEN
    float acc = cb[c];
#pragma unroll
    for (int k = 0; k < D_CONV; ++k) {
        int ls = l - (D_CONV - 1) + k;
        if (ls >= 0)
            acc += cw[c * D_CONV + k] * zx[(brow + ls) * (long)D_IN_PROJ + D_INNER + c];
    }
    xconv[row * (long)CONV_DIM + c] = silu_f(acc);
}

// dt = softplus(raw + dt_bias)
__global__ void dt_softplus_kernel(const float* __restrict__ zx, const float* __restrict__ dt_bias,
                                   float* __restrict__ dtb) {
    long i = (long)blockIdx.x * blockDim.x + threadIdx.x;
    if (i >= (long)NTOK * NHEADS) return;
    int  h   = (int)(i % NHEADS);
    long row = i / NHEADS;
    dtb[i] = softplus_f(zx[row * (long)D_IN_PROJ + D_INNER + CONV_DIM + h] + dt_bias[h]);
}

// ---------------------------------------------------------------------------
// SSD sequential scan. One block per (head, batch). State S[64x64] lives in
// registers: thread t owns p = t>>2 and 16 states n = (t&3)*16 .. +15.
// 64-timestep chunks staged into LDS (async if available); y reduced with
// wave32 shuffles.
// ---------------------------------------------------------------------------
#define SCAN_CHUNK 64

__global__ __launch_bounds__(256)
void ssd_scan_kernel(const float* __restrict__ xconv, const float* __restrict__ dtb,
                     const float* __restrict__ A_log, float* __restrict__ ys) {
    const int h = blockIdx.x;
    const int b = blockIdx.y;
    const int tid = threadIdx.x;
    const int p  = tid >> 2;
    const int nb = (tid & 3) * 16;
    const float A = -__expf(A_log[h]);

    __shared__ __align__(16) float xs[SCAN_CHUNK * HEADDIM];
    __shared__ __align__(16) float Bs[SCAN_CHUNK * D_STATE];
    __shared__ __align__(16) float Cs[SCAN_CHUNK * D_STATE];
    __shared__ float dts[SCAN_CHUNK];
    __shared__ float dAs[SCAN_CHUNK];

    float S[16];
#pragma unroll
    for (int j = 0; j < 16; ++j) S[j] = 0.f;

    for (int c0 = 0; c0 < SEQLEN; c0 += SCAN_CHUNK) {
        __syncthreads();
        // Stage chunk: 4096 floats per array = 1024 16B-chunks, 4 per thread.
#pragma unroll
        for (int it = 0; it < 4; ++it) {
            int li  = tid + it * 256;          // 0..1023
            int tt  = li >> 4;                 // 0..63
            int col = (li & 15) * 4;           // 0..60
            long row = (long)b * SEQLEN + c0 + tt;
            const float* base = xconv + row * (long)CONV_DIM;
#ifdef USE_ASYNC_LDS
            async_copy16(base + h * HEADDIM + col,           &xs[tt * HEADDIM + col]);
            async_copy16(base + D_INNER + col,               &Bs[tt * D_STATE + col]);
            async_copy16(base + D_INNER + D_STATE + col,     &Cs[tt * D_STATE + col]);
#else
            *reinterpret_cast<float4*>(&xs[tt * HEADDIM + col]) =
                *reinterpret_cast<const float4*>(base + h * HEADDIM + col);
            *reinterpret_cast<float4*>(&Bs[tt * D_STATE + col]) =
                *reinterpret_cast<const float4*>(base + D_INNER + col);
            *reinterpret_cast<float4*>(&Cs[tt * D_STATE + col]) =
                *reinterpret_cast<const float4*>(base + D_INNER + D_STATE + col);
#endif
        }
        if (tid < SCAN_CHUNK) {
            long row = (long)b * SEQLEN + c0 + tid;
            float dtv = dtb[row * NHEADS + h];
            dts[tid] = dtv;
            dAs[tid] = __expf(dtv * A);
        }
        ASYNC_WAIT();
        __syncthreads();

        for (int tt = 0; tt < SCAN_CHUNK; ++tt) {
            float dtv  = dts[tt];
            float dAv  = dAs[tt];
            float coef = dtv * xs[tt * HEADDIM + p];
            float yp = 0.f;
#pragma unroll
            for (int j = 0; j < 16; ++j) {
                float Bv = Bs[tt * D_STATE + nb + j];
                float Cv = Cs[tt * D_STATE + nb + j];
                S[j] = S[j] * dAv + coef * Bv;
                yp  += S[j] * Cv;
            }
            // Reduce 4 partial sums per p (contiguous lanes in one wave32).
            yp += __shfl_xor(yp, 1, 32);
            yp += __shfl_xor(yp, 2, 32);
            if ((tid & 3) == 0)
                ys[((long)b * SEQLEN + c0 + tt) * (long)D_INNER + h * HEADDIM + p] = yp;
        }
    }
}

// ---------------------------------------------------------------------------
// Fused: y = (ys + Dp*xh) * silu(z); RMSNorm over D_INNER; * norm_w; -> bf16
// One block of 256 threads per token (8 elems/thread).
// ---------------------------------------------------------------------------
__global__ __launch_bounds__(256)
void gate_rmsnorm_kernel(const float* __restrict__ ys, const float* __restrict__ xconv,
                         const float* __restrict__ zx, const float* __restrict__ Dp,
                         const float* __restrict__ nw, __bf16* __restrict__ out) {
    long row = blockIdx.x;
    int  tid = threadIdx.x;
    float vals[8];
    float ss = 0.f;
#pragma unroll
    for (int j = 0; j < 8; ++j) {
        int d = tid + j * 256;
        int h = d >> 6;
        float yv = ys[row * (long)D_INNER + d] + Dp[h] * xconv[row * (long)CONV_DIM + d];
        float zv = zx[row * (long)D_IN_PROJ + d];
        yv *= silu_f(zv);
        vals[j] = yv;
        ss += yv * yv;
    }
#pragma unroll
    for (int m = 16; m >= 1; m >>= 1) ss += __shfl_xor(ss, m, 32);
    __shared__ float red[8];
    __shared__ float scale_s;
    if ((tid & 31) == 0) red[tid >> 5] = ss;
    __syncthreads();
    if (tid == 0) {
        float t = 0.f;
#pragma unroll
        for (int w = 0; w < 8; ++w) t += red[w];
        scale_s = rsqrtf(t / (float)D_INNER + EPS_RMS);
    }
    __syncthreads();
    float sc = scale_s;
#pragma unroll
    for (int j = 0; j < 8; ++j) {
        int d = tid + j * 256;
        out[row * (long)D_INNER + d] = (__bf16)(vals[j] * sc * nw[d]);
    }
}

// out = y_fwd + flip_L(y_bwd)
__global__ void combine_kernel(const float* __restrict__ yf, const float* __restrict__ yb,
                               float* __restrict__ out) {
    long i = (long)blockIdx.x * blockDim.x + threadIdx.x;
    if (i >= (long)NTOK * D_MODEL) return;
    int  d   = (int)(i % D_MODEL);
    long row = i / D_MODEL;
    int  l   = (int)(row % SEQLEN);
    long b0  = row - l;
    out[i] = yf[i] + yb[(b0 + (SEQLEN - 1 - l)) * (long)D_MODEL + d];
}

// ---------------------------------------------------------------------------
// Host launcher
// ---------------------------------------------------------------------------
extern "C" void kernel_launch(void* const* d_in, const int* in_sizes, int n_in,
                              void* d_out, int out_size, void* d_ws, size_t ws_size,
                              hipStream_t stream) {
    (void)in_sizes; (void)n_in; (void)out_size; (void)ws_size;

    const float* x = (const float*)d_in[0];
    const float* in_w[2]    = {(const float*)d_in[1],  (const float*)d_in[9]};
    const float* conv_w[2]  = {(const float*)d_in[2],  (const float*)d_in[10]};
    const float* conv_b[2]  = {(const float*)d_in[3],  (const float*)d_in[11]};
    const float* dt_bias[2] = {(const float*)d_in[4],  (const float*)d_in[12]};
    const float* A_log[2]   = {(const float*)d_in[5],  (const float*)d_in[13]};
    const float* Dp[2]      = {(const float*)d_in[6],  (const float*)d_in[14]};
    const float* norm_w[2]  = {(const float*)d_in[7],  (const float*)d_in[15]};
    const float* out_w[2]   = {(const float*)d_in[8],  (const float*)d_in[16]};

    char* ws = (char*)d_ws;
    size_t off = 0;
    auto alloc = [&](size_t bytes) -> char* {
        char* p = ws + off;
        off += (bytes + 255) & ~(size_t)255;
        return p;
    };

    __bf16* xbf[2];
    xbf[0]          = (__bf16*)alloc((size_t)NTOK * D_MODEL * 2);
    xbf[1]          = (__bf16*)alloc((size_t)NTOK * D_MODEL * 2);
    __bf16* inw_bf  = (__bf16*)alloc((size_t)D_IN_PROJ * D_MODEL * 2); // reused per dir
    __bf16* outw_bf = (__bf16*)alloc((size_t)D_MODEL * D_INNER * 2);   // reused per dir
    float*  zx      = (float*)alloc((size_t)NTOK * D_IN_PROJ * 4);     // reused per dir
    float*  xconv   = (float*)alloc((size_t)NTOK * CONV_DIM * 4);      // reused per dir
    float*  dtb     = (float*)alloc((size_t)NTOK * NHEADS * 4);        // reused per dir
    float*  ysc     = (float*)alloc((size_t)NTOK * D_INNER * 4);       // reused per dir
    __bf16* ybf     = (__bf16*)alloc((size_t)NTOK * D_INNER * 2);      // reused per dir
    float*  yproj[2];
    yproj[0]        = (float*)alloc((size_t)NTOK * D_MODEL * 4);
    yproj[1]        = (float*)alloc((size_t)NTOK * D_MODEL * 4);

    const int TPB = 256;
    auto nblk = [](long n) { return (unsigned)((n + 255) / 256); };

    // x converts (forward order + L-flipped for backward direction)
    cvt_bf16_kernel<<<nblk((long)NTOK * D_MODEL), TPB, 0, stream>>>(x, xbf[0], (long)NTOK * D_MODEL);
    cvt_bf16_flipL_kernel<<<nblk((long)NTOK * D_MODEL), TPB, 0, stream>>>(x, xbf[1]);

    for (int dir = 0; dir < 2; ++dir) {
        // Weight converts (buffer reused across directions; stream-ordered)
        cvt_bf16_kernel<<<nblk((long)D_IN_PROJ * D_MODEL), TPB, 0, stream>>>(
            in_w[dir], inw_bf, (long)D_IN_PROJ * D_MODEL);
        cvt_bf16_kernel<<<nblk((long)D_MODEL * D_INNER), TPB, 0, stream>>>(
            out_w[dir], outw_bf, (long)D_MODEL * D_INNER);

        // GEMM1: zxbcdt[8192,4256] = x @ in_w^T  (K = 1024)
        {
            dim3 grid(D_IN_PROJ / GEMM_BN, NTOK / GEMM_BM);
            gemm_bf16_wmma_kernel<<<grid, TPB, 0, stream>>>(xbf[dir], inw_bf, zx,
                                                            NTOK, D_IN_PROJ, D_MODEL);
        }

        conv_silu_kernel<<<nblk((long)NTOK * CONV_DIM), TPB, 0, stream>>>(
            zx, conv_w[dir], conv_b[dir], xconv);
        dt_softplus_kernel<<<nblk((long)NTOK * NHEADS), TPB, 0, stream>>>(
            zx, dt_bias[dir], dtb);

        {
            dim3 grid(NHEADS, BATCH);
            ssd_scan_kernel<<<grid, TPB, 0, stream>>>(xconv, dtb, A_log[dir], ysc);
        }

        gate_rmsnorm_kernel<<<NTOK, TPB, 0, stream>>>(ysc, xconv, zx, Dp[dir], norm_w[dir], ybf);

        // GEMM2: yproj[8192,1024] = y @ out_w^T  (K = 2048)
        {
            dim3 grid(D_MODEL / GEMM_BN, NTOK / GEMM_BM);
            gemm_bf16_wmma_kernel<<<grid, TPB, 0, stream>>>(ybf, outw_bf, yproj[dir],
                                                            NTOK, D_MODEL, D_INNER);
        }
    }

    combine_kernel<<<nblk((long)NTOK * D_MODEL), TPB, 0, stream>>>(yproj[0], yproj[1],
                                                                   (float*)d_out);
}
